// MLA_11888469475378
// MI455X (gfx1250) — matrix operand order repaired and checked by reading, not compiled, not run on hardware
//
#include <hip/hip_runtime.h>

typedef __attribute__((ext_vector_type(16))) __bf16 v16bf;
typedef __attribute__((ext_vector_type(8)))  float  v8f;

union Frag { v16bf v; uint4 u[2]; __bf16 h[16]; };
union H8   { uint4 u; __bf16 h[8]; };

#define WMMA_BF16(A, B, C) \
  __builtin_amdgcn_wmma_f32_16x16x32_bf16(false, (A), false, (B), (short)0, (C), false, false)

// ---------------------------------------------------------------------------
// Generic tiled GEMM: C[M,N] = A[M,K] @ B[K,N], double-buffered LDS.
//   A: fp32 (converted to bf16 at stage) or bf16
//   B: always fp32 (original weights), converted + transposed into LDS
//   C: bf16 (intermediates) or fp32 (final output)
// Block = 256 threads = 8 waves; wave grid 4(M) x 2(N); wave tile 32x64.
// ---------------------------------------------------------------------------
#define BM 128
#define BN 128
#define BK 32
#define BKP 40   // padded LDS stride (halfwords) to break bank conflicts

template<typename AT>
__device__ __forceinline__ void gemm_load_stage(
    const AT* __restrict__ A, const float* __restrict__ B,
    int K, int N, int bm, int bn, int k0,
    int ar, int ac, int bk, int bnb, __bf16* ta, __bf16* tb)
{
  const AT* srcA = A + (size_t)(bm + ar) * K + (k0 + ac);
  if constexpr (sizeof(AT) == 4) {
    #pragma unroll
    for (int j = 0; j < 4; ++j) {
      float4 f = *(const float4*)((const float*)srcA + j * 4);
      ta[j*4+0] = (__bf16)f.x; ta[j*4+1] = (__bf16)f.y;
      ta[j*4+2] = (__bf16)f.z; ta[j*4+3] = (__bf16)f.w;
    }
  } else {
    const uint4* s4 = (const uint4*)srcA;
    ((uint4*)ta)[0] = s4[0];
    ((uint4*)ta)[1] = s4[1];
  }
  const float* srcB = B + (size_t)(k0 + bk) * N + (bn + bnb);
  #pragma unroll
  for (int j = 0; j < 4; ++j) {
    float4 f = make_float4(0.f, 0.f, 0.f, 0.f);
    if (bn + bnb + j * 4 + 4 <= N) f = *(const float4*)(srcB + j * 4);
    tb[j*4+0] = (__bf16)f.x; tb[j*4+1] = (__bf16)f.y;
    tb[j*4+2] = (__bf16)f.z; tb[j*4+3] = (__bf16)f.w;
  }
}

__device__ __forceinline__ void gemm_store_stage(
    __bf16* As, __bf16* Bs, int ar, int ac, int bk, int bnb,
    const __bf16* ta, const __bf16* tb)
{
  *(uint4*)&As[ar * BKP + ac]     = ((const uint4*)ta)[0];
  *(uint4*)&As[ar * BKP + ac + 8] = ((const uint4*)ta)[1];
  #pragma unroll
  for (int j = 0; j < 16; ++j)
    Bs[(bnb + j) * BKP + bk] = tb[j];
}

template<typename AT, typename OT>
__global__ __launch_bounds__(256) void gemm_bf16_wmma(
    const AT* __restrict__ A, const float* __restrict__ B, OT* __restrict__ C,
    int M, int N, int K)
{
  __shared__ __align__(16) __bf16 As[2][BM * BKP];
  __shared__ __align__(16) __bf16 Bs[2][BN * BKP];

  const int tid  = threadIdx.x;
  const int lane = tid & 31;
  const int w    = tid >> 5;
  const int wm   = w >> 1;           // 0..3
  const int wn   = w & 1;            // 0..1
  const int bm   = blockIdx.y * BM;
  const int bn   = blockIdx.x * BN;
  const int col  = lane & 15;
  const int sA   = (lane < 16) ? 0 : 8;    // A-fragment K sub-offset
  const int sB   = (lane < 16) ? 0 : 8;    // B-fragment K sub-offset

  v8f acc[2][4] = {};

  const int ar  = tid >> 1;            // A tile row 0..127
  const int ac  = (tid & 1) * 16;      // A tile col base {0,16}
  const int bk  = tid >> 3;            // B tile k-row 0..31
  const int bnb = (tid & 7) * 16;      // B tile col base 0..112

  __bf16 ta[16], tb[16];
  gemm_load_stage<AT>(A, B, K, N, bm, bn, 0, ar, ac, bk, bnb, ta, tb);
  gemm_store_stage(As[0], Bs[0], ar, ac, bk, bnb, ta, tb);
  __syncthreads();

  const int nIter = K / BK;
  for (int i = 0; i < nIter; ++i) {
    const int cur = i & 1;
    // stage next tile's global loads while WMMAs run on the current buffer
    if (i + 1 < nIter)
      gemm_load_stage<AT>(A, B, K, N, bm, bn, (i + 1) * BK,
                          ar, ac, bk, bnb, ta, tb);
    // ---- batch-load ALL fragments, then 8 independent WMMAs ----
    Frag a[2], bb[4];
    #pragma unroll
    for (int tm = 0; tm < 2; ++tm) {
      const __bf16* p = &As[cur][(wm * 32 + tm * 16 + col) * BKP];
      a[tm].u[0] = *(const uint4*)(p + sA);
      a[tm].u[1] = *(const uint4*)(p + 16 + sA);
    }
    #pragma unroll
    for (int tn = 0; tn < 4; ++tn) {
      const __bf16* p = &Bs[cur][(wn * 64 + tn * 16 + col) * BKP + sB];
      bb[tn].u[0] = *(const uint4*)p;
      bb[tn].u[1] = *(const uint4*)(p + 16);
    }
    #pragma unroll
    for (int tn = 0; tn < 4; ++tn)
      #pragma unroll
      for (int tm = 0; tm < 2; ++tm)
        acc[tm][tn] = WMMA_BF16(a[tm].v, bb[tn].v, acc[tm][tn]);
    // commit staged tile into the alternate buffer; one barrier per step
    if (i + 1 < nIter) {
      gemm_store_stage(As[cur ^ 1], Bs[cur ^ 1], ar, ac, bk, bnb, ta, tb);
      __syncthreads();
    }
  }
  // ---- store C ----
  #pragma unroll
  for (int tm = 0; tm < 2; ++tm) {
    const int rowb = bm + wm * 32 + tm * 16 + ((lane < 16) ? 0 : 8);
    #pragma unroll
    for (int tn = 0; tn < 4; ++tn) {
      const int c = bn + wn * 64 + tn * 16 + col;
      if (c < N) {
        #pragma unroll
        for (int r = 0; r < 8; ++r)
          C[(size_t)(rowb + r) * N + c] = (OT)acc[tm][tn][r];
      }
    }
  }
}

// ---------------------------------------------------------------------------
// RoPE + head-major assembly:
//   qh[b][h][l][0:128]=q_content, [128:192]=rope(q_rope)
//   kh[b][h][l][0:128]=k_content, [128:192]=rope(k_rope) (broadcast over h)
//   vh[b][h][l][0:128]=v
// ---------------------------------------------------------------------------
__global__ __launch_bounds__(256) void mla_assemble(
    const __bf16* __restrict__ qc, const __bf16* __restrict__ qr,
    const __bf16* __restrict__ kc, const __bf16* __restrict__ kr,
    const __bf16* __restrict__ vv,
    __bf16* __restrict__ qh, __bf16* __restrict__ kh, __bf16* __restrict__ vh)
{
  const int bl  = blockIdx.x;          // 0..4095 == b*2048 + pos
  const int b   = bl >> 11;
  const int pos = bl & 2047;
  for (int idx = threadIdx.x; idx < 16 * 192; idx += 256) {
    const int hh = idx / 192;
    const int d  = idx - hh * 192;
    __bf16 qv, kv;
    if (d < 128) {
      qv = qc[(size_t)bl * 2048 + hh * 128 + d];
      kv = kc[(size_t)bl * 2048 + hh * 128 + d];
    } else {
      const int j = d - 128;               // 0..63
      const int p = j & 31;
      // angle = pos * 10000^(-2p/64)
      const float ang = (float)pos * __expf(-(float)(2 * p) * (9.210340371976184f / 64.f));
      float sn, cs;
      __sincosf(ang, &sn, &cs);
      const int jo = (j < 32) ? (j + 32) : (j - 32);
      const float xj = (float)qr[(size_t)bl * 1024 + hh * 64 + j];
      const float xo = (float)qr[(size_t)bl * 1024 + hh * 64 + jo];
      qv = (__bf16)((j < 32) ? (xj * cs - xo * sn) : (xj * cs + xo * sn));
      const float yj = (float)kr[(size_t)bl * 64 + j];
      const float yo = (float)kr[(size_t)bl * 64 + jo];
      kv = (__bf16)((j < 32) ? (yj * cs - yo * sn) : (yj * cs + yo * sn));
    }
    const size_t dst = ((size_t)(b * 16 + hh) * 2048 + pos) * 192 + d;
    qh[dst] = qv;
    kh[dst] = kv;
  }
  for (int idx = threadIdx.x; idx < 16 * 128; idx += 256) {
    const int hh = idx >> 7;
    const int d  = idx & 127;
    vh[((size_t)(b * 16 + hh) * 2048 + pos) * 128 + d] =
        vv[(size_t)bl * 2048 + idx];
  }
}

// ---------------------------------------------------------------------------
// Flash attention (causal), one (b, h, 128-q-row tile) per block.
// 8 waves x 16 q-rows; 64-key LDS tiles; online softmax in C/D layout.
// Next K/V tile prefetched (global_prefetch_b8) while WMMAs run.
// ---------------------------------------------------------------------------
#define KSTR 200   // padded K LDS stride (elems)
#define VSTR 72    // padded V^T / P LDS strides (elems)

__global__ __launch_bounds__(256) void mla_flash_attn(
    const __bf16* __restrict__ Qh, const __bf16* __restrict__ Kh,
    const __bf16* __restrict__ Vh, __bf16* __restrict__ Oh)
{
  __shared__ __align__(16) __bf16 Ks[64 * KSTR];       // [key][192] padded
  __shared__ __align__(16) __bf16 Vs[128 * VSTR];      // [col][key] padded
  __shared__ __align__(16) __bf16 Ps[8 * 16 * VSTR];   // per-wave [16][64] padded

  const int tid  = threadIdx.x;
  const int lane = tid & 31;
  const int w    = tid >> 5;
  const int qt   = blockIdx.x;
  const int h    = blockIdx.y;
  const int b    = blockIdx.z;
  const __bf16* Q  = Qh + (size_t)(b * 16 + h) * 2048 * 192;
  const __bf16* Kp = Kh + (size_t)(b * 16 + h) * 2048 * 192;
  const __bf16* Vp = Vh + (size_t)(b * 16 + h) * 2048 * 128;
  const int col   = lane & 15;
  const int sA    = (lane < 16) ? 0 : 8;
  const int sB    = (lane < 16) ? 0 : 8;
  const int half8 = sA;
  const int qrow    = qt * 128 + w * 16 + col;
  const int rowbase = qt * 128 + w * 16 + half8;
  const float scale = 0.07216878364870323f;   // 1/sqrt(192)

  // Q fragments (6 chunks of K=32) loaded once, straight from global.
  Frag qf[6];
  #pragma unroll
  for (int c = 0; c < 6; ++c) {
    const __bf16* p = Q + (size_t)qrow * 192 + c * 32;
    qf[c].u[0] = *(const uint4*)(p + sA);
    qf[c].u[1] = *(const uint4*)(p + 16 + sA);
  }

  v8f o[8] = {};
  float m[8], l[8];
  #pragma unroll
  for (int r = 0; r < 8; ++r) { m[r] = -1e30f; l[r] = 0.f; }

  const int krow = tid >> 2;          // 0..63 (K & V tile loaders)
  const int kcb  = (tid & 3) * 48;    // K col base 0..144
  const int vnb  = (tid & 3) * 32;    // V col base 0,32,64,96
  const int ntiles = qt * 2 + 2;

  for (int it = 0; it < ntiles; ++it) {
    const int kt = it * 64;
    __syncthreads();
    // ---- K tile -> LDS row-major (padded) ----
    #pragma unroll
    for (int j = 0; j < 6; ++j)
      *(uint4*)&Ks[krow * KSTR + kcb + j * 8] =
          *(const uint4*)(Kp + (size_t)(kt + krow) * 192 + kcb + j * 8);
    // ---- V tile -> LDS transposed [col][key] (padded) ----
    #pragma unroll
    for (int j = 0; j < 4; ++j) {
      H8 vt;
      vt.u = *(const uint4*)(Vp + (size_t)(kt + krow) * 128 + vnb + j * 8);
      #pragma unroll
      for (int e = 0; e < 8; ++e)
        Vs[(vnb + j * 8 + e) * VSTR + krow] = vt.h[e];
    }
    __syncthreads();
    // ---- prefetch next tile while this one is consumed ----
    if (it + 1 < ntiles) {
      __builtin_prefetch(Kp + (size_t)(kt + 64 + krow) * 192 + kcb, 0, 0);
      __builtin_prefetch(Vp + (size_t)(kt + 64 + krow) * 128 + vnb, 0, 0);
    }

    // ---- S = Q @ K^T : per chunk, batch 4 B-frags then 4 indep WMMAs ----
    v8f s[4] = {};
    #pragma unroll
    for (int c = 0; c < 6; ++c) {
      Frag kb[4];
      #pragma unroll
      for (int t = 0; t < 4; ++t) {
        const __bf16* p = &Ks[(t * 16 + col) * KSTR + c * 32 + sB];
        kb[t].u[0] = *(const uint4*)p;
        kb[t].u[1] = *(const uint4*)(p + 16);
      }
      #pragma unroll
      for (int t = 0; t < 4; ++t)
        s[t] = WMMA_BF16(qf[c].v, kb[t].v, s[t]);
    }
    // ---- scale + causal mask + tile row-max ----
    float mt[8];
    #pragma unroll
    for (int r = 0; r < 8; ++r) mt[r] = -1e30f;
    #pragma unroll
    for (int t = 0; t < 4; ++t) {
      const int kcol = kt + t * 16 + col;
      #pragma unroll
      for (int r = 0; r < 8; ++r) {
        float v = s[t][r] * scale;
        if (kcol > rowbase + r) v = -1e30f;
        s[t][r] = v;
        mt[r] = fmaxf(mt[r], v);
      }
    }
    #pragma unroll
    for (int mask = 1; mask <= 8; mask <<= 1) {
      #pragma unroll
      for (int r = 0; r < 8; ++r)
        mt[r] = fmaxf(mt[r], __shfl_xor(mt[r], mask, 32));
    }
    // ---- online softmax update ----
    float rs[8];
    #pragma unroll
    for (int r = 0; r < 8; ++r) {
      const float mn = fmaxf(m[r], mt[r]);
      const float f  = __expf(m[r] - mn);
      m[r] = mn;
      l[r] *= f;
      #pragma unroll
      for (int t2 = 0; t2 < 8; ++t2) o[t2][r] = o[t2][r] * f;
      rs[r] = 0.f;
    }
    #pragma unroll
    for (int t = 0; t < 4; ++t) {
      #pragma unroll
      for (int r = 0; r < 8; ++r) {
        const float p = __expf(s[t][r] - m[r]);
        rs[r] += p;
        Ps[w * (16 * VSTR) + (r + half8) * VSTR + t * 16 + col] = (__bf16)p;
      }
    }
    #pragma unroll
    for (int mask = 1; mask <= 8; mask <<= 1) {
      #pragma unroll
      for (int r = 0; r < 8; ++r)
        rs[r] += __shfl_xor(rs[r], mask, 32);
    }
    #pragma unroll
    for (int r = 0; r < 8; ++r) l[r] += rs[r];

    // ---- O += P @ V : batch 4 V-frags then 4 indep WMMAs ----
    #pragma unroll
    for (int c = 0; c < 2; ++c) {
      Frag pa;
      const __bf16* p = &Ps[w * (16 * VSTR) + col * VSTR + c * 32];
      pa.u[0] = *(const uint4*)(p + sA);
      pa.u[1] = *(const uint4*)(p + 16 + sA);
      #pragma unroll
      for (int g = 0; g < 2; ++g) {
        Frag vb[4];
        #pragma unroll
        for (int t = 0; t < 4; ++t) {
          const __bf16* vp = &Vs[((g * 4 + t) * 16 + col) * VSTR + c * 32 + sB];
          vb[t].u[0] = *(const uint4*)vp;
          vb[t].u[1] = *(const uint4*)(vp + 16);
        }
        #pragma unroll
        for (int t = 0; t < 4; ++t)
          o[g * 4 + t] = WMMA_BF16(pa.v, vb[t].v, o[g * 4 + t]);
      }
    }
  }
  // ---- epilogue: normalize, store bf16 [B*L][NH*HD] ----
  #pragma unroll
  for (int r = 0; r < 8; ++r) {
    const float inv = (l[r] > 0.f) ? (1.f / l[r]) : 0.f;
    #pragma unroll
    for (int t2 = 0; t2 < 8; ++t2)
      Oh[(size_t)(b * 2048 + rowbase + r) * 2048 + h * 128 + t2 * 16 + col] =
          (__bf16)(o[t2][r] * inv);
  }
}

// ---------------------------------------------------------------------------
extern "C" void kernel_launch(void* const* d_in, const int* in_sizes, int n_in,
                              void* d_out, int out_size, void* d_ws, size_t ws_size,
                              hipStream_t stream)
{
  (void)in_sizes; (void)n_in; (void)out_size; (void)ws_size;
  const float* x        = (const float*)d_in[0];
  const float* wq_down  = (const float*)d_in[1];
  const float* wq_up    = (const float*)d_in[2];
  const float* wq_rope  = (const float*)d_in[3];
  const float* wkv_down = (const float*)d_in[4];
  const float* wk_up    = (const float*)d_in[5];
  const float* wv_up    = (const float*)d_in[6];
  const float* wk_rope  = (const float*)d_in[7];
  const float* wo       = (const float*)d_in[8];
  float* out = (float*)d_out;

  char* ws = (char*)d_ws;
  size_t off = 0;
  auto alloc = [&](size_t elems) {
    size_t o = off;
    off += ((elems * 2 + 255) & ~(size_t)255);
    return o;
  };
  const size_t M = 4096;                      // B*L
  const size_t o_qlat  = alloc(M * 768);
  const size_t o_kvlat = alloc(M * 512);
  const size_t o_qc    = alloc(M * 2048);     // reused as attn_out later
  const size_t o_qr    = alloc(M * 1024);
  const size_t o_kc    = alloc(M * 2048);
  const size_t o_vv    = alloc(M * 2048);
  const size_t o_kr    = alloc(M * 64);
  const size_t o_qh    = alloc((size_t)2 * 16 * 2048 * 192);
  const size_t o_kh    = alloc((size_t)2 * 16 * 2048 * 192);
  const size_t o_vh    = alloc((size_t)2 * 16 * 2048 * 128);

  __bf16* qlat  = (__bf16*)(ws + o_qlat);
  __bf16* kvlat = (__bf16*)(ws + o_kvlat);
  __bf16* qc    = (__bf16*)(ws + o_qc);
  __bf16* qr    = (__bf16*)(ws + o_qr);
  __bf16* kc    = (__bf16*)(ws + o_kc);
  __bf16* vv    = (__bf16*)(ws + o_vv);
  __bf16* kr    = (__bf16*)(ws + o_kr);
  __bf16* qh    = (__bf16*)(ws + o_qh);
  __bf16* kh    = (__bf16*)(ws + o_kh);
  __bf16* vh    = (__bf16*)(ws + o_vh);
  __bf16* attn  = (__bf16*)(ws + o_qc);       // alias: qc dead after assemble

  const dim3 blk(256);
  auto grd = [](int N) { return dim3((N + 127) / 128, 32); };

  // down-projections + rope key base (A = fp32 x)
  gemm_bf16_wmma<float,  __bf16><<<grd(768),  blk, 0, stream>>>(x,     wq_down,  qlat, 4096,  768, 2048);
  gemm_bf16_wmma<float,  __bf16><<<grd(512),  blk, 0, stream>>>(x,     wkv_down, kvlat,4096,  512, 2048);
  gemm_bf16_wmma<float,  __bf16><<<grd(64),   blk, 0, stream>>>(x,     wk_rope,  kr,   4096,   64, 2048);
  // up-projections (A = bf16 latents)
  gemm_bf16_wmma<__bf16, __bf16><<<grd(2048), blk, 0, stream>>>(qlat,  wq_up,    qc,   4096, 2048,  768);
  gemm_bf16_wmma<__bf16, __bf16><<<grd(1024), blk, 0, stream>>>(qlat,  wq_rope,  qr,   4096, 1024,  768);
  gemm_bf16_wmma<__bf16, __bf16><<<grd(2048), blk, 0, stream>>>(kvlat, wk_up,    kc,   4096, 2048,  512);
  gemm_bf16_wmma<__bf16, __bf16><<<grd(2048), blk, 0, stream>>>(kvlat, wv_up,    vv,   4096, 2048,  512);
  // rope + head-major layout
  mla_assemble<<<dim3(4096), blk, 0, stream>>>(qc, qr, kc, kr, vv, qh, kh, vh);
  // causal flash attention
  mla_flash_attn<<<dim3(16, 16, 2), blk, 0, stream>>>(qh, kh, vh, attn);
  // output projection (f32 out)
  gemm_bf16_wmma<__bf16, float><<<grd(2048), blk, 0, stream>>>(attn, wo, out, 4096, 2048, 2048);
}
